// TimeLSTM_30623116820668
// MI455X (gfx1250) — compile-verified, matrix operand order checked by reading, not run on hardware
//
#include <hip/hip_runtime.h>
#include <hip/hip_bf16.h>

// ---------------- problem dims ----------------
#define BB 128
#define SS 512
#define II 256
#define HH 512
#define G4H 2048          // 4*H
#define KC  768           // concat K = H + I

// ---------------- decomposition ----------------
#define NJ 16             // H slices
#define NBT 8             // batch tiles
#define NWG (NJ*NBT)      // 128 persistent workgroups
#define JT 32             // hidden cols per workgroup
#define BT 16             // batch rows per workgroup
#define NTHREADS 320      // 10 waves; 1 wave per 16x16 WMMA tile

// ---------------- workspace layout (bytes) ----------------
#define WCAT_OFF 0u                       // [2048][768] bf16 = 3145728
#define WD_OFF   3145728u                 // [512][512]  bf16 = 524288
#define XBF_OFF  3670016u                 // [128][512][256] bf16 = 33554432
#define BALL_OFF 37224448u                // [2048] f32 (W_all_b + U_all_b)
#define BD_OFF   37232640u                // [512]  f32
#define HBF_OFF  37234688u                // 2 x [128][512] bf16 ping-pong = 262144
#define CBF_OFF  37496832u                // 2 x [128][512] bf16 ping-pong = 262144
#define SYNC_OFF 37758976u                // [512] int barrier counters
#define ZERO_WORDS 131584u                // (HBF+CBF+SYNC) bytes / 4

// ---------------- LDS layout (bytes) ----------------
#define L_WG  0u          // [128][768] bf16 gate+U weight slice   196608
#define L_WD  196608u     // [32][512]  bf16 W_d slice              32768
#define L_AH  229376u     // [16][512]  bf16 h tile                 16384
#define L_AC  245760u     // [16][512]  bf16 c tile                 16384
#define L_AX  262144u     // [16][256]  bf16 x tile                  8192
#define L_ACC 270336u     // [10][16][16] f32 wmma results          10240
#define L_C   280576u     // [16][32] f32 resident cell state        2048
#define L_BG  282624u     // [128] f32 gate bias slice                512
#define L_BD  283136u     // [32]  f32 d bias slice                   128
#define LDS_BYTES 283264u // 276.6 KB < 320 KB/WGP

typedef __attribute__((ext_vector_type(4)))  float   v4f;
typedef __attribute__((ext_vector_type(8)))  float   v8f;
typedef __attribute__((ext_vector_type(4)))  __bf16  v4bf;
typedef __attribute__((ext_vector_type(8)))  __bf16  v8bf;
typedef __attribute__((ext_vector_type(16))) __bf16  v16bf;
typedef __attribute__((ext_vector_type(4)))  unsigned int u32x4;
typedef __attribute__((ext_vector_type(4)))  int     i32x4;
typedef __attribute__((ext_vector_type(8)))  int     i32x8;

// ---- Tensor Data Mover availability (this toolchain: 6-arg builtin) ----
#if defined(__has_builtin)
#if __has_builtin(__builtin_amdgcn_tensor_load_to_lds) && __has_builtin(__builtin_amdgcn_s_wait_tensorcnt)
#define USE_TDM 1
#endif
#endif
#ifndef USE_TDM
#define USE_TDM 0
#endif

__device__ __forceinline__ v16bf ld2x8(const __bf16* p0, const __bf16* p1) {
  v8bf lo = *(const v8bf*)p0;
  v8bf hi = *(const v8bf*)p1;
  v16bf r;
#pragma unroll
  for (int i = 0; i < 8; ++i) { r[i] = lo[i]; r[i + 8] = hi[i]; }
  return r;
}

__device__ __forceinline__ float sigm(float x) {
  return 1.0f / (1.0f + __expf(-x));
}
__device__ __forceinline__ float ftanh(float x) {
  x = fminf(fmaxf(x, -15.0f), 15.0f);
  float e = __expf(2.0f * x);
  return (e - 1.0f) / (e + 1.0f);
}

#if USE_TDM
// Build a D# (groups 0-3) for a 2D bf16 tile load (tile_d1==0 -> 1D) and issue it.
// All arguments are workgroup-uniform -> descriptor lands in SGPRs.
// clang-23 6-arg form: (u32x4 g0, i32x8 g1, i32x4 g2, i32x4 g3, i32x8 extra, i32 cpol)
__device__ __forceinline__ void tdm_load_2d(unsigned lds_off, const void* gptr,
                                            unsigned tensor_d0, unsigned tensor_d1,
                                            unsigned tile_d0, unsigned tile_d1,
                                            unsigned long long stride0)
{
  unsigned long long ga = (unsigned long long)gptr;
  u32x4 g0;
  g0[0] = 1u;                                           // count=1, user descriptor
  g0[1] = lds_off;                                      // LDS byte address
  g0[2] = (unsigned)ga;                                 // global addr [31:0]
  g0[3] = (unsigned)((ga >> 32) & 0x01FFFFFFu)          // global addr [56:32]
        | (2u << 30);                                   // type=2 ("image")
  i32x8 g1;
  g1[0] = (int)(1u << 16);                              // data_size=1 -> 2 bytes
  g1[1] = (int)((tensor_d0 & 0xFFFFu) << 16);           // tensor_dim0[15:0]
  g1[2] = (int)(((tensor_d0 >> 16) & 0xFFFFu)
        | ((tensor_d1 & 0xFFFFu) << 16));               // dim0[31:16] | dim1[15:0]
  g1[3] = (int)(((tensor_d1 >> 16) & 0xFFFFu)
        | ((tile_d0 & 0xFFFFu) << 16));                 // dim1[31:16] | tile_dim0
  g1[4] = (int)(tile_d1 & 0xFFFFu);                     // tile_dim1 | tile_dim2=0
  g1[5] = (int)(unsigned)(stride0 & 0xFFFFFFFFull);     // dim0_stride[31:0]
  g1[6] = (int)(unsigned)((stride0 >> 32) & 0xFFFFull); // dim0_stride[47:32]
  g1[7] = 0;                                            // dim1_stride = 0
  i32x4 g2 = {0, 0, 0, 0};                              // >2D groups unused
  i32x4 g3 = {0, 0, 0, 0};
  i32x8 g4 = {0, 0, 0, 0, 0, 0, 0, 0};
  __builtin_amdgcn_tensor_load_to_lds(g0, g1, g2, g3, g4, 0);
}
#endif

// ========== prep: pack [W_all | U_all] -> bf16, W_d -> bf16, fuse biases ==========
__global__ void prep_weights(const float* __restrict__ Wall, const float* __restrict__ Uall,
                             const float* __restrict__ Wdw,  const float* __restrict__ Wallb,
                             const float* __restrict__ Uallb,const float* __restrict__ Wdb,
                             __bf16* __restrict__ wcat, __bf16* __restrict__ wd,
                             float* __restrict__ ball, float* __restrict__ bd)
{
  const size_t NW = (size_t)G4H * KC;          // 1572864
  const size_t ND = (size_t)HH * HH;           // 262144
  size_t i = (size_t)blockIdx.x * blockDim.x + threadIdx.x;
  if (i < NW) {
    size_t n = i / KC, k = i % KC;
    float v = (k < HH) ? Wall[n * HH + k] : Uall[n * II + (k - HH)];
    wcat[i] = (__bf16)v;
  } else if (i < NW + ND) {
    size_t k = i - NW;
    wd[k] = (__bf16)Wdw[k];
  } else if (i < NW + ND + G4H) {
    size_t n = i - NW - ND;
    ball[n] = Wallb[n] + Uallb[n];              // fold U bias into gate bias
  } else if (i < NW + ND + G4H + HH) {
    size_t n = i - NW - ND - G4H;
    bd[n] = Wdb[n];
  }
}

__global__ void prep_inputs(const float* __restrict__ x, __bf16* __restrict__ xbf, size_t n4)
{
  size_t i = (size_t)blockIdx.x * blockDim.x + threadIdx.x;
  if (i < n4) {
    v4f v = *(const v4f*)(x + i * 4);
    v4bf o;
#pragma unroll
    for (int k = 0; k < 4; ++k) o[k] = (__bf16)v[k];
    *(v4bf*)(xbf + i * 4) = o;
  }
}

__global__ void init_state(unsigned int* __restrict__ p, size_t n)
{
  size_t i = (size_t)blockIdx.x * blockDim.x + threadIdx.x;
  if (i < n) p[i] = 0u;
}

// ========== persistent scan kernel ==========
__global__ __launch_bounds__(NTHREADS, 1)
void timelstm_scan(const __bf16* __restrict__ xbf,
                   const __bf16* __restrict__ wcat_g,
                   const __bf16* __restrict__ wd_g,
                   const float*  __restrict__ ball_g,
                   const float*  __restrict__ bd_g,
                   const float*  __restrict__ ts,     // [B,S]
                   __bf16* hbf, __bf16* cbf,          // 2 x [B,H] ping-pong each
                   int* sync_g,
                   float* __restrict__ out)           // [B,S,H]
{
  extern __shared__ char smem[];
  __bf16* Wg   = (__bf16*)(smem + L_WG);
  __bf16* Wd   = (__bf16*)(smem + L_WD);
  __bf16* Ah   = (__bf16*)(smem + L_AH);
  __bf16* Ac   = (__bf16*)(smem + L_AC);
  __bf16* Ax   = (__bf16*)(smem + L_AX);
  float*  AccT = (float*)(smem + L_ACC);
  float*  Cst  = (float*)(smem + L_C);
  float*  Bg   = (float*)(smem + L_BG);
  float*  Bd   = (float*)(smem + L_BD);

  const int tid = threadIdx.x;
  const int wg  = blockIdx.x;
  const int bt  = wg >> 4;            // 0..7 batch tile
  const int js  = wg & 15;            // 0..15 hidden slice
  const int b0  = bt * BT;
  const int j0  = js * JT;

  // ---- one-time: stage weight slices into LDS (resident for all 512 steps) ----
  {
    const uint4* src = (const uint4*)wcat_g;          // 96 uint4 per 768-elem row
    uint4* dst = (uint4*)Wg;
    for (int i = tid; i < 128 * 96; i += NTHREADS) {
      int r = i / 96, c = i % 96;
      int n = (r >> 5) * HH + j0 + (r & 31);          // gate-major local rows
      dst[r * 96 + c] = src[(size_t)n * 96 + c];
    }
    const uint4* srcd = (const uint4*)wd_g;           // 64 uint4 per 512-elem row
    uint4* dstd = (uint4*)Wd;
    for (int i = tid; i < 32 * 64; i += NTHREADS) {
      int r = i >> 6, c = i & 63;
      dstd[r * 64 + c] = srcd[(size_t)(j0 + r) * 64 + c];
    }
    if (tid < 128)                Bg[tid] = ball_g[(tid >> 5) * HH + j0 + (tid & 31)];
    if (tid >= 128 && tid < 160)  Bd[tid - 128] = bd_g[j0 + (tid - 128)];
    for (int e = tid; e < BT * JT; e += NTHREADS) Cst[e] = 0.0f;
  }
  __syncthreads();

  const int wv   = tid >> 5;          // wave 0..9 -> one 16x16 output tile each
  const int lane = tid & 31;
  const int lm   = lane & 15;
  const int half = lane >> 4;

  for (int s = 0; s < SS; ++s) {
    const int rb = s & 1, wb = rb ^ 1;
    const __bf16* hsrc = hbf + rb * (BB * HH);
    const __bf16* csrc = cbf + rb * (BB * HH);

    // ---- stage A operands (h, c, x_s rows of this batch tile) into LDS ----
#if USE_TDM
    if (wv == 0) {
      // h, c: 16 contiguous rows of 512 bf16 -> single 1D 16 KB DMA each
      tdm_load_2d(L_AH, hsrc + (size_t)b0 * HH, BT * HH, 1, BT * HH, 0, BT * HH);
      tdm_load_2d(L_AC, csrc + (size_t)b0 * HH, BT * HH, 1, BT * HH, 0, BT * HH);
      // x_s: 16 rows x 256 bf16, row stride = S*I elements
      tdm_load_2d(L_AX, xbf + ((size_t)b0 * SS + s) * II,
                  II, BT, II, BT, (unsigned long long)SS * II);
      __builtin_amdgcn_s_wait_tensorcnt(0);
    }
#else
    {
      const uint4* hs = (const uint4*)hsrc;
      const uint4* cs = (const uint4*)csrc;
      const uint4* xs = (const uint4*)xbf;
      uint4* ah4 = (uint4*)Ah; uint4* ac4 = (uint4*)Ac; uint4* ax4 = (uint4*)Ax;
      for (int i = tid; i < 2560; i += NTHREADS) {
        if (i < 1024) {                               // h: 16 rows x 64 uint4
          int r = i >> 6, c = i & 63;
          ah4[i] = hs[(size_t)(b0 + r) * 64 + c];
        } else if (i < 2048) {                        // c
          int k = i - 1024; int r = k >> 6, c = k & 63;
          ac4[k] = cs[(size_t)(b0 + r) * 64 + c];
        } else {                                      // x_s: 16 rows x 32 uint4
          int k = i - 2048; int r = k >> 5, c = k & 31;
          ax4[k] = xs[((size_t)(b0 + r) * SS + s) * 32 + c];
        }
      }
    }
#endif
    if (s + 1 < SS) {                                 // warm L2 for next x slice
      for (int i = tid; i < 64; i += NTHREADS) {
        int r = i >> 2, c = i & 3;
        __builtin_prefetch(xbf + ((size_t)(b0 + r) * SS + (s + 1)) * II + c * 64, 0, 1);
      }
    }
    __syncthreads();

    // ---- WMMA: wave wv owns one 16x16 tile ----
    v8f acc = {0.f, 0.f, 0.f, 0.f, 0.f, 0.f, 0.f, 0.f};
    if (wv < 8) {
      // gate tiles: gate g = wv>>1, col half jb; A = [h | x], K = 768
      const int g  = wv >> 1;
      const int jb = (wv & 1) * 16;
      const __bf16* ah = Ah + lm * HH;
      const __bf16* ax = Ax + lm * II;
      const __bf16* bw = Wg + (size_t)(g * 32 + jb + lm) * KC;
#pragma unroll
      for (int kk = 0; kk < HH; kk += 32) {
        v16bf a = ld2x8(ah + kk + half * 8, ah + kk + 16 + half * 8);
        v16bf b = ld2x8(bw + kk + half * 8, bw + kk + 16 + half * 8);
        acc = __builtin_amdgcn_wmma_f32_16x16x32_bf16(false, a, false, b,
                                                      (short)0, acc, false, false);
      }
#pragma unroll
      for (int kk = 0; kk < II; kk += 32) {
        v16bf a = ld2x8(ax + kk + half * 8, ax + kk + 16 + half * 8);
        v16bf b = ld2x8(bw + HH + kk + half * 8, bw + HH + kk + 16 + half * 8);
        acc = __builtin_amdgcn_wmma_f32_16x16x32_bf16(false, a, false, b,
                                                      (short)0, acc, false, false);
      }
    } else {
      // decay tiles: c @ W_d.T, K = 512
      const int jb = (wv - 8) * 16;
      const __bf16* ac = Ac + lm * HH;
      const __bf16* bw = Wd + (size_t)(jb + lm) * HH;
#pragma unroll
      for (int kk = 0; kk < HH; kk += 32) {
        v16bf a = ld2x8(ac + kk + half * 8, ac + kk + 16 + half * 8);
        v16bf b = ld2x8(bw + kk + half * 8, bw + kk + 16 + half * 8);
        acc = __builtin_amdgcn_wmma_f32_16x16x32_bf16(false, a, false, b,
                                                      (short)0, acc, false, false);
      }
    }
    {
      // C/D layout: lanes 0-15 N=lane, vgpr r -> M = r + 8*half
      float* tp = AccT + wv * 256;
#pragma unroll
      for (int r = 0; r < 8; ++r) tp[(r + half * 8) * 16 + lm] = acc[r];
    }
    __syncthreads();

    // ---- elementwise gate fusion on the owned [16 x 32] state slice ----
    for (int e = tid; e < BT * JT; e += NTHREADS) {
      int m = e >> 5, j = e & 31;
      int jhi = j >> 4, jl = j & 15;
      float pf = AccT[(0 + jhi) * 256 + m * 16 + jl] + Bg[j];
      float pi = AccT[(2 + jhi) * 256 + m * 16 + jl] + Bg[32 + j];
      float po = AccT[(4 + jhi) * 256 + m * 16 + jl] + Bg[64 + j];
      float pc = AccT[(6 + jhi) * 256 + m * 16 + jl] + Bg[96 + j];
      float ps = AccT[(8 + jhi) * 256 + m * 16 + jl] + Bd[j];
      int   b  = b0 + m;
      float tv    = ts[(size_t)b * SS + s];
      float cs1   = ftanh(ps);
      float c_old = Cst[e];
      float c_adj = c_old - cs1 + cs1 * tv;
      float f  = sigm(pf), ii = sigm(pi), o = sigm(po), ct = sigm(pc);
      float c_new = f * c_adj + ii * ct;
      float h_new = o * ftanh(c_new);
      Cst[e] = c_new;                                  // f32 state stays local
      int jg = j0 + j;
      out[((size_t)b * SS + s) * HH + jg] = h_new;     // f32 output
      hbf[wb * (BB * HH) + b * HH + jg] = (__bf16)h_new;  // publish bf16 for GEMMs
      cbf[wb * (BB * HH) + b * HH + jg] = (__bf16)c_new;
    }

    // ---- grid-wide barrier (L2 atomics, one per step) ----
    __threadfence();
    __syncthreads();
    if (s + 1 < SS) {
      if (tid == 0) {
        __hip_atomic_fetch_add(&sync_g[s], 1, __ATOMIC_ACQ_REL, __HIP_MEMORY_SCOPE_AGENT);
        while (__hip_atomic_load(&sync_g[s], __ATOMIC_ACQUIRE, __HIP_MEMORY_SCOPE_AGENT) < NWG) {
          __builtin_amdgcn_s_sleep(1);
        }
      }
      __syncthreads();
      __threadfence();
    }
  }
}

extern "C" void kernel_launch(void* const* d_in, const int* in_sizes, int n_in,
                              void* d_out, int out_size, void* d_ws, size_t ws_size,
                              hipStream_t stream) {
  const float* inputs = (const float*)d_in[0];   // [B,S,I]
  const float* ts     = (const float*)d_in[1];   // [B,S]
  const float* Wall   = (const float*)d_in[2];   // [4H,H]
  const float* Wallb  = (const float*)d_in[3];   // [4H]
  const float* Uall   = (const float*)d_in[4];   // [4H,I]
  const float* Uallb  = (const float*)d_in[5];   // [4H]
  const float* Wdw    = (const float*)d_in[6];   // [H,H]
  const float* Wdb    = (const float*)d_in[7];   // [H]
  float* out = (float*)d_out;                    // [B,S,H]

  char* ws = (char*)d_ws;
  __bf16* wcat = (__bf16*)(ws + WCAT_OFF);
  __bf16* wd   = (__bf16*)(ws + WD_OFF);
  __bf16* xbf  = (__bf16*)(ws + XBF_OFF);
  float*  ball = (float*)(ws + BALL_OFF);
  float*  bd   = (float*)(ws + BD_OFF);
  __bf16* hbf  = (__bf16*)(ws + HBF_OFF);
  __bf16* cbf  = (__bf16*)(ws + CBF_OFF);
  int*    sync = (int*)(ws + SYNC_OFF);

  // 1) weights -> bf16 (concat [W_all | U_all]), fused biases
  size_t nprep = (size_t)G4H * KC + (size_t)HH * HH + G4H + HH;
  prep_weights<<<(unsigned)((nprep + 255) / 256), 256, 0, stream>>>(
      Wall, Uall, Wdw, Wallb, Uallb, Wdb, wcat, wd, ball, bd);

  // 2) inputs -> bf16
  size_t n4 = (size_t)BB * SS * II / 4;
  prep_inputs<<<(unsigned)((n4 + 255) / 256), 256, 0, stream>>>(inputs, xbf, n4);

  // 3) zero h/c ping-pong state and barrier counters (every launch: graph-safe)
  init_state<<<(ZERO_WORDS + 255) / 256, 256, 0, stream>>>(
      (unsigned int*)(ws + HBF_OFF), (size_t)ZERO_WORDS);

  // 4) persistent fused scan: 128 workgroups (1/WGP, 276KB LDS each), 10 waves
  timelstm_scan<<<NWG, NTHREADS, LDS_BYTES, stream>>>(
      xbf, wcat, wd, ball, bd, ts, hbf, cbf, sync, out);
}
